// SRGNN_Pooling_47794396070380
// MI455X (gfx1250) — matrix operand order repaired and checked by reading, not compiled, not run on hardware
//
#include <hip/hip_runtime.h>
#include <math.h>

// ---------------------------------------------------------------------------
// SRGNN pooling for MI455X (gfx1250, wave32, WMMA).
// Key optimization: local_rep_repeat @ W1 == (local_rep @ W1)[batch], so the
// N-row W1 GEMM collapses to a B-row GEMM + gather (65.5 -> 2.1 GFLOP).
// Remaining big GEMM (x @ W2, 65.5 GFLOP) runs as split-precision bf16 WMMA
// (hi/lo, 3 products, f32 accumulate) to hit the matrix-core rate while
// keeping ~fp32 accuracy. HBM floor: x read 3x = 1.5 GB ~ 65us @ 23.3 TB/s.
// ---------------------------------------------------------------------------

typedef __attribute__((ext_vector_type(16))) __bf16 v16bf;
typedef __attribute__((ext_vector_type(8)))  float  v8f;

#define NN 500000
#define DD 256
#define BB 16384

// ---- 3-product split-precision bf16 WMMA: (Ah+Al)*(Bh+Bl) ~= AhBh+AhBl+AlBh
static __device__ __forceinline__ v8f wmma3(v16bf ah, v16bf al,
                                            v16bf bh, v16bf bl, v8f acc) {
  acc = __builtin_amdgcn_wmma_f32_16x16x32_bf16(false, al, false, bh,
                                                (short)0, acc, false, false);
  acc = __builtin_amdgcn_wmma_f32_16x16x32_bf16(false, ah, false, bl,
                                                (short)0, acc, false, false);
  acc = __builtin_amdgcn_wmma_f32_16x16x32_bf16(false, ah, false, bh,
                                                (short)0, acc, false, false);
  return acc;
}

// Convert 16 f32 (two 8-element runs) into bf16 hi + lo-residual A operand.
static __device__ __forceinline__ void cvt_a(const float* p0, const float* p1,
                                             v16bf& ah, v16bf& al) {
  alignas(16) float v[16];
  *(float4*)(v + 0)  = *(const float4*)(p0);
  *(float4*)(v + 4)  = *(const float4*)(p0 + 4);
  *(float4*)(v + 8)  = *(const float4*)(p1);
  *(float4*)(v + 12) = *(const float4*)(p1 + 4);
#pragma unroll
  for (int j = 0; j < 16; ++j) {
    __bf16 h = (__bf16)v[j];
    ah[j] = h;
    al[j] = (__bf16)(v[j] - (float)h);
  }
}

// ---------------------------------------------------------------------------
// Kernel: segment start offsets from sorted batch[] (seg_start[BB] = N).
__global__ void seg_starts_kernel(const int* __restrict__ batch,
                                  int* __restrict__ seg_start) {
  int i = blockIdx.x * blockDim.x + threadIdx.x;
  if (i >= NN) return;
  int b = batch[i];
  int prev = (i == 0) ? -1 : batch[i - 1];
  for (int s = prev + 1; s <= b; ++s) seg_start[s] = i;
  if (i == NN - 1) {
    for (int s = b + 1; s <= BB; ++s) seg_start[s] = NN;
  }
}

// ---------------------------------------------------------------------------
// Kernel: pre-pack a K x Ncols f32 weight matrix into the WMMA B-operand
// lane layout (ISA 7.12.2: lane group 0 holds K=0..7,16..23, group 1 K=8..15,24..31 of
// each 32-K tile; lane%16 selects column within a 16-wide column tile),
// split into bf16 hi + residual lo.
__global__ void prepack_kernel(const float* __restrict__ W, int K, int Ncols,
                               __bf16* __restrict__ Bh, __bf16* __restrict__ Bl) {
  int idx = blockIdx.x * blockDim.x + threadIdx.x;
  if (idx >= K * Ncols) return;
  int j    = idx & 15;
  int L    = (idx >> 4) & 31;
  int tile = idx >> 9;
  int KT   = K >> 5;
  int kt   = tile % KT;
  int ct   = tile / KT;
  int g    = L >> 4;
  int n    = (L & 15) + ct * 16;
  int kk   = kt * 32 + ((j < 8) ? (g * 8 + j) : (16 + g * 8 + (j - 8)));
  float w  = W[(size_t)kk * Ncols + n];
  __bf16 h = (__bf16)w;
  Bh[idx] = h;
  Bl[idx] = (__bf16)(w - (float)h);
}

// ---------------------------------------------------------------------------
// Kernel: per-segment weighted column sum: out[b][t] = sum_i x[i][t] * w[i].
// One block per segment, 256 threads = 256 columns, fully coalesced.
__global__ void pool_kernel(const float* __restrict__ x,
                            const float* __restrict__ w,
                            const int* __restrict__ seg_start,
                            float* __restrict__ out) {
  int b = blockIdx.x;
  int t = threadIdx.x;
  int s = seg_start[b], e = seg_start[b + 1];
  float acc = 0.f;
  for (int i = s; i < e; ++i) acc += x[(size_t)i * DD + t] * w[i];
  out[(size_t)b * DD + t] = acc;
}

// ---------------------------------------------------------------------------
// Kernel: generic M x K @ K x 256 WMMA GEMM with optional concat A = [A0|A1]
// and bias(es). One wave computes a 16-row x 64-col (4 column-tile) slab.
__global__ void __launch_bounds__(256)
gemm16_kernel(const float* __restrict__ A0, const float* __restrict__ A1, int K,
              const __bf16* __restrict__ Bh, const __bf16* __restrict__ Bl,
              const float* __restrict__ bias0, const float* __restrict__ bias1,
              float* __restrict__ out, int M) {
  int wave = threadIdx.x >> 5;
  int lane = threadIdx.x & 31;
  int task = blockIdx.x * (blockDim.x >> 5) + wave;
  int ntasks = (M >> 4) * 4;
  if (task >= ntasks) return;
  int rt  = task >> 2;
  int ctg = task & 3;
  int r0  = rt << 4;
  int g   = lane >> 4;
  int nl  = lane & 15;
  int KT  = K >> 5;

  v8f acc[4];
#pragma unroll
  for (int c = 0; c < 4; ++c) acc[c] = (v8f)0.f;

  for (int kt = 0; kt < KT; ++kt) {
    int row = r0 + nl;
    int k0 = kt * 32 + g * 8;
    int k1 = kt * 32 + 16 + g * 8;
    // 8-element runs start at multiples of 8 so they never cross the DD
    // boundary of the concatenated [A0|A1] operand.
    const float* p0 = (k0 < DD) ? (A0 + (size_t)row * DD + k0)
                                : (A1 + (size_t)row * DD + (k0 - DD));
    const float* p1 = (k1 < DD) ? (A0 + (size_t)row * DD + k1)
                                : (A1 + (size_t)row * DD + (k1 - DD));
    v16bf ah, al;
    cvt_a(p0, p1, ah, al);
#pragma unroll
    for (int c = 0; c < 4; ++c) {
      int ct = ctg * 4 + c;
      size_t boff = ((size_t)(ct * KT + kt) * 32 + lane) * 16;
      v16bf bh = *(const v16bf*)(Bh + boff);
      v16bf bl = *(const v16bf*)(Bl + boff);
      acc[c] = wmma3(ah, al, bh, bl, acc[c]);
    }
  }

#pragma unroll
  for (int c = 0; c < 4; ++c) {
    int col = (ctg * 4 + c) * 16 + nl;
    float bs = bias0[col] + (bias1 ? bias1[col] : 0.f);
#pragma unroll
    for (int r = 0; r < 8; ++r) {
      int m = r + g * 8;  // C/D layout: lanes 16-31 hold rows 8..15
      out[(size_t)(r0 + m) * DD + col] = acc[c][r] + bs;
    }
  }
}

// ---------------------------------------------------------------------------
// Kernel: fused attention. One wave per 16-row x-tile:
//   att[i] = sigmoid( (x @ W2)[i] + hloc[batch[i]] ) . w3
// where hloc = local_rep @ W1 + b1 + b2 was precomputed per segment.
// x tile held resident in VGPRs as bf16 hi/lo (128 VGPRs).
__global__ void __launch_bounds__(64)
attention_kernel(const float* __restrict__ x, const int* __restrict__ batch,
                 const float* __restrict__ hloc,
                 const __bf16* __restrict__ W2h, const __bf16* __restrict__ W2l,
                 const float* __restrict__ w3, float* __restrict__ att) {
  int wave = threadIdx.x >> 5;
  int lane = threadIdx.x & 31;
  int tile = blockIdx.x * (blockDim.x >> 5) + wave;
  if (tile >= NN / 16) return;
  int r0 = tile << 4;
  int g  = lane >> 4;
  int nl = lane & 15;
  const int KT = DD / 32;  // 8

  // Load + convert the whole 16x256 x tile (A operand, all 8 k-tiles).
  v16bf ah[8], al[8];
  {
    const float* rp = x + (size_t)(r0 + nl) * DD;
#pragma unroll
    for (int kt = 0; kt < 8; ++kt)
      cvt_a(rp + kt * 32 + g * 8, rp + kt * 32 + 16 + g * 8, ah[kt], al[kt]);
  }

  int sb[8];
#pragma unroll
  for (int r = 0; r < 8; ++r) sb[r] = batch[r0 + g * 8 + r];

  float partial[8];
#pragma unroll
  for (int r = 0; r < 8; ++r) partial[r] = 0.f;

  for (int ct = 0; ct < 16; ++ct) {
    v8f acc = (v8f)0.f;
#pragma unroll
    for (int kt = 0; kt < 8; ++kt) {
      size_t boff = ((size_t)(ct * KT + kt) * 32 + lane) * 16;
      v16bf bh = *(const v16bf*)(W2h + boff);
      v16bf bl = *(const v16bf*)(W2l + boff);
      acc = wmma3(ah[kt], al[kt], bh, bl, acc);
    }
    int col = ct * 16 + nl;
    float w3v = w3[col];
#pragma unroll
    for (int r = 0; r < 8; ++r) {
      float h = hloc[(size_t)sb[r] * DD + col];  // gather: includes b1+b2
      float z = acc[r] + h;
      float s = 1.f / (1.f + __expf(-z));
      partial[r] += s * w3v;
    }
  }

  // Reduce the 16 column-lanes per half-wave (keep lane groups separate).
#pragma unroll
  for (int m = 1; m < 16; m <<= 1) {
#pragma unroll
    for (int r = 0; r < 8; ++r) partial[r] += __shfl_xor(partial[r], m, 32);
  }
  if (nl == 0) {
#pragma unroll
    for (int r = 0; r < 8; ++r) att[r0 + g * 8 + r] = partial[r];
  }
}

// ---------------------------------------------------------------------------
extern "C" void kernel_launch(void* const* d_in, const int* in_sizes, int n_in,
                              void* d_out, int out_size, void* d_ws, size_t ws_size,
                              hipStream_t stream) {
  (void)in_sizes; (void)n_in; (void)out_size; (void)ws_size;
  const float* x    = (const float*)d_in[0];
  const int*   batch= (const int*)d_in[1];
  const float* mask = (const float*)d_in[2];
  // d_in[3] = num_segments scalar (compile-time BB)
  const float* W1 = (const float*)d_in[4];
  const float* b1 = (const float*)d_in[5];
  const float* W2 = (const float*)d_in[6];
  const float* b2 = (const float*)d_in[7];
  const float* w3 = (const float*)d_in[8];
  const float* W4 = (const float*)d_in[9];
  const float* b4 = (const float*)d_in[10];
  float* out = (float*)d_out;

  // Workspace carve-up (~51 MB total).
  char* ws = (char*)d_ws;
  size_t off = 0;
  auto carve = [&](size_t bytes) -> char* {
    char* p = ws + off;
    off = (off + bytes + 255) & ~(size_t)255;
    return p;
  };
  float*  local_rep  = (float*)carve((size_t)BB * DD * 4);
  float*  global_rep = (float*)carve((size_t)BB * DD * 4);
  float*  hloc       = (float*)carve((size_t)BB * DD * 4);
  float*  att        = (float*)carve((size_t)NN * 4);
  int*    seg_start  = (int*)carve((size_t)(BB + 1) * 4);
  __bf16* W1h = (__bf16*)carve((size_t)DD * DD * 2);
  __bf16* W1l = (__bf16*)carve((size_t)DD * DD * 2);
  __bf16* W2h = (__bf16*)carve((size_t)DD * DD * 2);
  __bf16* W2l = (__bf16*)carve((size_t)DD * DD * 2);
  __bf16* W4h = (__bf16*)carve((size_t)2 * DD * DD * 2);
  __bf16* W4l = (__bf16*)carve((size_t)2 * DD * DD * 2);

  // 1. segment offsets (batch is sorted)
  seg_starts_kernel<<<(NN + 255) / 256, 256, 0, stream>>>(batch, seg_start);
  // 2. weight pre-pack into WMMA B layout, bf16 hi/lo
  prepack_kernel<<<(DD * DD + 255) / 256, 256, 0, stream>>>(W1, DD, DD, W1h, W1l);
  prepack_kernel<<<(DD * DD + 255) / 256, 256, 0, stream>>>(W2, DD, DD, W2h, W2l);
  prepack_kernel<<<(2 * DD * DD + 255) / 256, 256, 0, stream>>>(W4, 2 * DD, DD, W4h, W4l);
  // 3. local_rep = segment_sum(x * mask)
  pool_kernel<<<BB, 256, 0, stream>>>(x, mask, seg_start, local_rep);
  // 4. hloc = local_rep @ W1 + b1 + b2   (B rows instead of N rows!)
  gemm16_kernel<<<((BB / 16) * 4) / 8, 256, 0, stream>>>(
      local_rep, nullptr, DD, W1h, W1l, b1, b2, hloc, BB);
  // 5. att = sigmoid(x @ W2 + hloc[batch]) @ w3   (fused, WMMA)
  attention_kernel<<<(NN / 16) / 2, 64, 0, stream>>>(x, batch, hloc, W2h, W2l,
                                                     w3, att);
  // 6. global_rep = segment_sum(x * att)
  pool_kernel<<<BB, 256, 0, stream>>>(x, att, seg_start, global_rep);
  // 7. out = [local_rep | global_rep] @ W4 + b4
  gemm16_kernel<<<((BB / 16) * 4) / 8, 256, 0, stream>>>(
      local_rep, global_rep, 2 * DD, W4h, W4l, b4, nullptr, out, BB);
}
